// Attention_26895085207812
// MI455X (gfx1250) — matrix operand order repaired and checked by reading, not compiled, not run on hardware
//
#include <hip/hip_runtime.h>

typedef __bf16 bf16_t;
typedef __attribute__((ext_vector_type(16))) __bf16 v16bf;
typedef __attribute__((ext_vector_type(8)))  __bf16 v8bf;
typedef __attribute__((ext_vector_type(8)))  float  v8f;

// ---------------------------------------------------------------------------
// WMMA fragment helpers (CDNA5 wave32, V_WMMA_F32_16X16X32_BF16)
// A 16x32: lane m=lane&15, half h=lane>>4; elems 0..7 -> k=k0+h*8+e,
//          elems 8..15 -> k=k0+16+h*8+(e-8)
// B 32x16: lane n=lane&15, half h; same k order as A
// C 16x16: lane n=lane&15; vgpr v -> m = h*8+v
// ---------------------------------------------------------------------------

__device__ __forceinline__ v8f vzero8() {
  v8f z = {0.f, 0.f, 0.f, 0.f, 0.f, 0.f, 0.f, 0.f};
  return z;
}

__device__ __forceinline__ v16bf combine16(v8bf lo, v8bf hi) {
  v16bf r;
#pragma unroll
  for (int e = 0; e < 8; ++e) { r[e] = lo[e]; r[e + 8] = hi[e]; }
  return r;
}

// A fragment from bf16 row-major [ld] (global or LDS pointer).
__device__ __forceinline__ v16bf load_a_bf16(const bf16_t* base, int ld,
                                             int m0, int k0, int lane) {
  const int m = lane & 15, h = (lane >> 4) & 1;
  const bf16_t* p0 = base + (size_t)(m0 + m) * ld + k0 + h * 8;
  v8bf lo = *(const v8bf*)p0;
  v8bf hi = *(const v8bf*)(p0 + 16);
  return combine16(lo, hi);
}

// B fragment from bf16 where rows are the output-N dimension (pre-transposed),
// i.e. element (k,n) lives at base[n*ld + k]; per-lane two contiguous 8-element runs.
__device__ __forceinline__ v16bf load_b_bf16(const bf16_t* base, int ld,
                                             int n0, int k0, int lane) {
  const int n = lane & 15, h = (lane >> 4) & 1;
  const bf16_t* p = base + (size_t)(n0 + n) * ld + k0 + h * 8;
  v8bf lo = *(const v8bf*)p;
  v8bf hi = *(const v8bf*)(p + 16);
  return combine16(lo, hi);
}

__device__ __forceinline__ v8f wmma_bf16(v16bf a, v16bf b, v8f c) {
  return __builtin_amdgcn_wmma_f32_16x16x32_bf16(
      /*neg_a=*/false, a, /*neg_b=*/false, b,
      /*c_mod=*/(short)0, c, /*reuse_a=*/false, /*reuse_b=*/false);
}

// ---------------------------------------------------------------------------
// Problem constants
// ---------------------------------------------------------------------------
#define BATCH 8
#define SEQ   1024
#define EMB   768
#define NH    12
#define HD    64
#define QKV3  2304  // 3*EMB
#define NTOK  8192  // BATCH*SEQ

// 32x64-per-wave GEMM building blocks --------------------------------------
__device__ __forceinline__ void load_frags(const bf16_t* __restrict__ A,
                                           const bf16_t* __restrict__ W,
                                           int m0, int c0, int k, int lane,
                                           v16bf a[2], v16bf b[4]) {
  a[0] = load_a_bf16(A, EMB, m0, k, lane);
  a[1] = load_a_bf16(A, EMB, m0 + 16, k, lane);
#pragma unroll
  for (int nt = 0; nt < 4; ++nt)
    b[nt] = load_b_bf16(W, EMB, c0 + nt * 16, k, lane);
}

__device__ __forceinline__ void mma8(const v16bf a[2], const v16bf b[4],
                                     v8f acc[2][4]) {
#pragma unroll
  for (int mi = 0; mi < 2; ++mi)
#pragma unroll
    for (int nt = 0; nt < 4; ++nt)
      acc[mi][nt] = wmma_bf16(a[mi], b[nt], acc[mi][nt]);
}

// ---------------------------------------------------------------------------
// Kernel 0: one-time converts. xb = bf16(x); wqkvT/woutT = bf16 transposes.
// ---------------------------------------------------------------------------
__global__ void prep_inputs(const float* __restrict__ x,
                            const float* __restrict__ w_qkv,
                            const float* __restrict__ w_out,
                            bf16_t* __restrict__ xb,
                            bf16_t* __restrict__ wqkvT,
                            bf16_t* __restrict__ woutT) {
  const int i = blockIdx.x * blockDim.x + threadIdx.x;
  const int NX = NTOK * EMB;       // 6,291,456
  const int NQ = EMB * QKV3;       // 1,769,472
  if (i < NX) {
    xb[i] = (__bf16)x[i];
  } else if (i < NX + NQ) {
    const int j = i - NX;
    const int c = j / EMB, k = j % EMB;
    wqkvT[j] = (__bf16)w_qkv[(size_t)k * QKV3 + c];
  } else {
    const int j = i - NX - NQ;
    if (j < EMB * EMB) {
      const int c = j / EMB, k = j % EMB;
      woutT[j] = (__bf16)w_out[(size_t)k * EMB + c];
    }
  }
}

// ---------------------------------------------------------------------------
// Kernel 1: QKV projection. One wave = 32x64 tile; k-loop unrolled x2 with
// ping-pong fragment buffers (no register swap code).
// Writes Q (scaled 1/sqrt(64)) and K as [BH,N,64], V transposed [BH,64,N].
// ---------------------------------------------------------------------------
__global__ __launch_bounds__(256) void qkv_gemm(const bf16_t* __restrict__ xb,
                                                const bf16_t* __restrict__ wqkvT,
                                                bf16_t* __restrict__ Qb,
                                                bf16_t* __restrict__ Kb,
                                                bf16_t* __restrict__ Vt) {
  const int lane = threadIdx.x & 31;
  const int wave = threadIdx.x >> 5;
  const int gid = blockIdx.x * 8 + wave;          // 9216 waves total
  const int NSTRIP = QKV3 / 64;                   // 36
  const int mt = gid / NSTRIP;                    // 0..255 (32-token tiles)
  const int strip = gid % NSTRIP;                 // 0..35
  const int m0 = mt * 32;
  const int c0 = strip * 64;

  v8f acc[2][4];
#pragma unroll
  for (int mi = 0; mi < 2; ++mi)
#pragma unroll
    for (int nt = 0; nt < 4; ++nt) acc[mi][nt] = vzero8();

  v16bf a0[2], b0[4], a1[2], b1[4];
  load_frags(xb, wqkvT, m0, c0, 0, lane, a0, b0);

  for (int k0 = 0; k0 < EMB; k0 += 64) {          // 12 iterations, no tail
    load_frags(xb, wqkvT, m0, c0, k0 + 32, lane, a1, b1);  // always valid
    mma8(a0, b0, acc);
    if (k0 + 64 < EMB)
      load_frags(xb, wqkvT, m0, c0, k0 + 64, lane, a0, b0);
    mma8(a1, b1, acc);
  }

  const int which = c0 / EMB;            // 0=Q 1=K 2=V (strips never cross)
  const int head = (c0 % EMB) / HD;      // strip == one head's 64 dims
  const int n = lane & 15, h = (lane >> 4) & 1;
#pragma unroll
  for (int mi = 0; mi < 2; ++mi) {
#pragma unroll
    for (int nt = 0; nt < 4; ++nt) {
#pragma unroll
      for (int v = 0; v < 8; ++v) {
        const int m = m0 + mi * 16 + h * 8 + v;   // global token
        const int bb = m >> 10, tn = m & 1023;
        const int d = nt * 16 + n;
        const int bh = bb * NH + head;
        const float val = acc[mi][nt][v];
        if (which == 0) {
          Qb[((size_t)bh * SEQ + tn) * HD + d] = (__bf16)(val * 0.125f);
        } else if (which == 1) {
          Kb[((size_t)bh * SEQ + tn) * HD + d] = (__bf16)val;
        } else {
          Vt[((size_t)bh * HD + d) * SEQ + tn] = (__bf16)val;
        }
      }
    }
  }
}

// ---------------------------------------------------------------------------
// Kernel 2: attention for one (b,h) and one 16-query tile per block.
// 128 threads (4 waves). Scores live in LDS as bf16 with padded row stride.
// ---------------------------------------------------------------------------
#define SLDB 1040  // 1024 + 16 bf16 pad -> 8-bank rotation per row

__global__ __launch_bounds__(128) void attn_kernel(const bf16_t* __restrict__ Qb,
                                                   const bf16_t* __restrict__ Kb,
                                                   const bf16_t* __restrict__ Vt,
                                                   bf16_t* __restrict__ O) {
  __shared__ bf16_t sc[16 * SLDB];  // ~33 KB

  const int lane = threadIdx.x & 31;
  const int wave = threadIdx.x >> 5;   // 0..3
  const int bh = blockIdx.x >> 6;      // 0..95
  const int qt = blockIdx.x & 63;      // 0..63
  const int q0 = qt * 16;

  const bf16_t* Qh = Qb + (size_t)bh * SEQ * HD;
  const bf16_t* Kh = Kb + (size_t)bh * SEQ * HD;
  const bf16_t* Vh = Vt + (size_t)bh * HD * SEQ;
  const int n = lane & 15, h = (lane >> 4) & 1;

  // ---- Phase 1: S = (Q/sqrt(d)) @ K^T, 16 x 1024, split over 4 waves ----
  {
    v16bf qa0 = load_a_bf16(Qh, HD, q0, 0, lane);
    v16bf qa1 = load_a_bf16(Qh, HD, q0, 32, lane);
    const int ktbase = wave * 16;
    v16bf e0, e1, o0, o1;  // even/odd key-tile ping-pong buffers
    e0 = load_b_bf16(Kh, HD, ktbase * 16, 0, lane);
    e1 = load_b_bf16(Kh, HD, ktbase * 16, 32, lane);
    for (int j = 0; j < 16; j += 2) {            // 8 iterations, no tail
      const int kt = ktbase + j;
      o0 = load_b_bf16(Kh, HD, (kt + 1) * 16, 0, lane);
      o1 = load_b_bf16(Kh, HD, (kt + 1) * 16, 32, lane);
      {
        v8f acc = vzero8();
        acc = wmma_bf16(qa0, e0, acc);
        acc = wmma_bf16(qa1, e1, acc);
#pragma unroll
        for (int v = 0; v < 8; ++v)
          sc[(size_t)(h * 8 + v) * SLDB + kt * 16 + n] = (__bf16)acc[v];
      }
      if (j + 2 < 16) {
        e0 = load_b_bf16(Kh, HD, (kt + 2) * 16, 0, lane);
        e1 = load_b_bf16(Kh, HD, (kt + 2) * 16, 32, lane);
      }
      {
        v8f acc = vzero8();
        acc = wmma_bf16(qa0, o0, acc);
        acc = wmma_bf16(qa1, o1, acc);
#pragma unroll
        for (int v = 0; v < 8; ++v)
          sc[(size_t)(h * 8 + v) * SLDB + (kt + 1) * 16 + n] = (__bf16)acc[v];
      }
    }
  }
  __syncthreads();

  // ---- Phase 2: row softmax; 8 lanes per row, wave-local shuffles ----
  {
    const int r = (wave << 2) | (lane >> 3);  // 0..15
    const int c = lane & 7;
    bf16_t* row = sc + (size_t)r * SLDB;
    float mx = -3.0e38f;
    for (int i = c; i < SEQ; i += 8) mx = fmaxf(mx, (float)row[i]);
    mx = fmaxf(mx, __shfl_xor(mx, 1, 8));
    mx = fmaxf(mx, __shfl_xor(mx, 2, 8));
    mx = fmaxf(mx, __shfl_xor(mx, 4, 8));
    float sum = 0.f;
    for (int i = c; i < SEQ; i += 8) {
      const float e = __expf((float)row[i] - mx);
      row[i] = (__bf16)e;
      sum += e;
    }
    sum += __shfl_xor(sum, 1, 8);
    sum += __shfl_xor(sum, 2, 8);
    sum += __shfl_xor(sum, 4, 8);
    const float inv = 1.0f / sum;
    for (int i = c; i < SEQ; i += 8) row[i] = (__bf16)((float)row[i] * inv);
  }
  __syncthreads();

  // ---- Phase 3: O_tile = P @ V ; each wave owns one 16-wide dim slice ----
  {
    const int nt = wave;  // 0..3 -> dims nt*16..nt*16+15
    v8f acc = vzero8();
    v16bf ae, be, ao, bo;
    ae = load_a_bf16(sc, SLDB, 0, 0, lane);           // LDS
    be = load_b_bf16(Vh, SEQ, nt * 16, 0, lane);      // global
    for (int k0 = 0; k0 < SEQ; k0 += 64) {            // 16 iterations, no tail
      ao = load_a_bf16(sc, SLDB, 0, k0 + 32, lane);
      bo = load_b_bf16(Vh, SEQ, nt * 16, k0 + 32, lane);
      acc = wmma_bf16(ae, be, acc);
      if (k0 + 64 < SEQ) {
        ae = load_a_bf16(sc, SLDB, 0, k0 + 64, lane);
        be = load_b_bf16(Vh, SEQ, nt * 16, k0 + 64, lane);
      }
      acc = wmma_bf16(ao, bo, acc);
    }
    const int bb = bh / NH, head = bh % NH;
#pragma unroll
    for (int v = 0; v < 8; ++v) {
      const int tok = bb * SEQ + q0 + h * 8 + v;
      const int col = head * HD + nt * 16 + n;
      O[(size_t)tok * EMB + col] = (__bf16)acc[v];
    }
  }
}

// ---------------------------------------------------------------------------
// Kernel 3: output projection  out = O @ w_out + b_out  (fp32 result)
// One wave = 32x64 tile; ping-pong k-loop.
// ---------------------------------------------------------------------------
__global__ __launch_bounds__(256) void out_gemm(const bf16_t* __restrict__ O,
                                                const bf16_t* __restrict__ woutT,
                                                const float* __restrict__ bias,
                                                float* __restrict__ out) {
  const int lane = threadIdx.x & 31;
  const int wave = threadIdx.x >> 5;
  const int gid = blockIdx.x * 8 + wave;  // 3072 waves total
  const int NSTRIP = EMB / 64;            // 12
  const int mt = gid / NSTRIP;            // 0..255
  const int strip = gid % NSTRIP;
  const int m0 = mt * 32;
  const int c0 = strip * 64;

  v8f acc[2][4];
#pragma unroll
  for (int mi = 0; mi < 2; ++mi)
#pragma unroll
    for (int nt = 0; nt < 4; ++nt) acc[mi][nt] = vzero8();

  v16bf a0[2], b0[4], a1[2], b1[4];
  load_frags(O, woutT, m0, c0, 0, lane, a0, b0);

  for (int k0 = 0; k0 < EMB; k0 += 64) {          // 12 iterations, no tail
    load_frags(O, woutT, m0, c0, k0 + 32, lane, a1, b1);
    mma8(a0, b0, acc);
    if (k0 + 64 < EMB)
      load_frags(O, woutT, m0, c0, k0 + 64, lane, a0, b0);
    mma8(a1, b1, acc);
  }

  const int n = lane & 15, h = (lane >> 4) & 1;
#pragma unroll
  for (int nt = 0; nt < 4; ++nt) {
    const int col = c0 + nt * 16 + n;
    const float bcol = bias[col];
#pragma unroll
    for (int mi = 0; mi < 2; ++mi) {
#pragma unroll
      for (int v = 0; v < 8; ++v) {
        const int m = m0 + mi * 16 + h * 8 + v;
        out[(size_t)m * EMB + col] = acc[mi][nt][v] + bcol;
      }
    }
  }
}

// ---------------------------------------------------------------------------
// Host launcher
// ---------------------------------------------------------------------------
extern "C" void kernel_launch(void* const* d_in, const int* in_sizes, int n_in,
                              void* d_out, int out_size, void* d_ws, size_t ws_size,
                              hipStream_t stream) {
  (void)in_sizes; (void)n_in; (void)out_size; (void)ws_size;
  const float* x      = (const float*)d_in[0];
  const float* w_qkv  = (const float*)d_in[1];
  const float* w_out  = (const float*)d_in[2];
  const float* b_out  = (const float*)d_in[3];
  float* out = (float*)d_out;

  char* ws = (char*)d_ws;
  // byte offsets (all 256-aligned)
  bf16_t* xb    = (bf16_t*)(ws + 0);          // 8192*768*2   = 12,582,912
  bf16_t* wqkvT = (bf16_t*)(ws + 12582912);   // 2304*768*2   =  3,538,944
  bf16_t* woutT = (bf16_t*)(ws + 16121856);   //  768*768*2   =  1,179,648
  bf16_t* Qb    = (bf16_t*)(ws + 17301504);   // 96*1024*64*2 = 12,582,912
  bf16_t* Kb    = (bf16_t*)(ws + 29884416);   // 12,582,912
  bf16_t* Vt    = (bf16_t*)(ws + 42467328);   // 12,582,912
  bf16_t* O     = (bf16_t*)(ws + 55050240);   // 12,582,912   (end 67,633,152)

  // 6,291,456 + 1,769,472 + 589,824 = 8,650,752 elements -> 33792 blocks
  prep_inputs<<<33792, 256, 0, stream>>>(x, w_qkv, w_out, xb, wqkvT, woutT);
  // 256 M-tiles * 36 strips = 9216 waves -> 1152 blocks of 8 waves
  qkv_gemm<<<1152, 256, 0, stream>>>(xb, wqkvT, Qb, Kb, Vt);
  // 96 heads * 64 query tiles
  attn_kernel<<<6144, 128, 0, stream>>>(Qb, Kb, Vt, O);
  // 256 M-tiles * 12 strips = 3072 waves -> 384 blocks of 8 waves
  out_gemm<<<384, 256, 0, stream>>>(O, woutT, b_out, out);
}
